// BidirectionalMamba3_81724637708617
// MI455X (gfx1250) — matrix operand order repaired and checked by reading, not compiled, not run on hardware
//
#include <hip/hip_runtime.h>
#include <hip/hip_bf16.h>
#include <math.h>

typedef __attribute__((ext_vector_type(16))) _Float16 v16h;
typedef __attribute__((ext_vector_type(8)))  float    v8f;
typedef __attribute__((ext_vector_type(4)))  unsigned u32x4;

#define D_MODEL   512
#define D_INNER   1024
#define D_STATE   64
#define NHEADS    16
#define HEADDIM   64
#define K_CONV    4
#define CONV_DIM  1152            // D_INNER + 2*D_STATE
#define D_IN_PROJ 2192            // 2*D_INNER + 2*D_STATE + NHEADS
#define BATCH     4
#define SEQLEN    1024
#define NROWS     (BATCH * SEQLEN) // 4096

// GEMM block tiling
#define MBLK 128
#define NBLK 64
#define KSTEP 32
#define LDS_ROW 80                      // 64B of K-data + 16B pad (bank-conflict free)
#define ABUF_BYTES (MBLK * LDS_ROW)     // 10240
#define BBUF_BYTES (NBLK * LDS_ROW)     // 5120
#define BUF_BYTES  (ABUF_BYTES + BBUF_BYTES)

// ---------------------------------------------------------------------------
// CDNA5 async global->LDS copy (ASYNCcnt-tracked, no VGPR round trip).
// The LDS destination is passed as a *pointer* (then truncated to the LDS
// byte offset: generic-pointer low 32 bits == LDS offset per the aperture
// rule). The ptrtoint captures the __shared__ object, so the "memory"
// clobber forces LLVM to keep the LDS fragment loads alive (passing a plain
// integer offset let the compiler prove the asm couldn't write LDS and it
// folded all ds_loads to undef).
// ---------------------------------------------------------------------------
__device__ __forceinline__ void async_g2l_b128(void* lds_ptr, const void* gaddr) {
  unsigned lds_off = (unsigned)(size_t)lds_ptr;
  asm volatile("global_load_async_to_lds_b128 %0, %1, off"
               :: "v"(lds_off), "v"(gaddr) : "memory");
}
__device__ __forceinline__ void wait_async0() {
#if __has_builtin(__builtin_amdgcn_s_wait_asynccnt)
  __builtin_amdgcn_s_wait_asynccnt(0);
#else
  asm volatile("s_wait_asynccnt 0" ::: "memory");
#endif
}

// ---------------------------------------------------------------------------
// fp32 -> fp16 conversion
// ---------------------------------------------------------------------------
__global__ void cvt_f32_f16_kernel(const float* __restrict__ src,
                                   _Float16* __restrict__ dst, int n) {
  int i = blockIdx.x * 256 + threadIdx.x;
  if (i < n) dst[i] = (_Float16)src[i];
}

// ---------------------------------------------------------------------------
// WMMA GEMM:  C[M,N] (f32) = A[M,K] (f16, row-major) * W[N,K]^T (f16, row-major)
// 256 threads / 8 waves per block; block computes a 128x64 tile.
// Per K-step: async-stage A(128x32) + B(64x32) into double-buffered LDS,
// each wave runs 4 x v_wmma_f32_16x16x32_f16 off LDS fragments.
// Fragment layouts used here (wave32; contiguous K = 16*(lane/16)+s for A and B, not the ISA table order):
//   A: lane = M%16; K pairs at (lane<16?0:16)
//   B: lane = N%16; K pairs at (lane<16?0:16)
//   C/D f32: VGPR r -> (M=r,N=lane) lane<16 ; (M=r+8,N=lane-16) else
// ---------------------------------------------------------------------------
__global__ void wmma_gemm_f16(const _Float16* __restrict__ A,
                              const _Float16* __restrict__ W,
                              float* __restrict__ C,
                              int M, int N, int K) {
  __shared__ __align__(16) char smem[2 * BUF_BYTES];
  const int tid  = threadIdx.x;
  const int lane = tid & 31;
  const int wave = tid >> 5;
  const int l15  = lane & 15;
  const int tilesN = (N + NBLK - 1) / NBLK;
  const int m0 = (blockIdx.x / tilesN) * MBLK;
  const int n0 = (blockIdx.x % tilesN) * NBLK;

  // ---- staging assignments (per thread: 2x A segments + 1x B segment, 16B each)
  const int crow0 = tid >> 2;          // 0..63
  const int cseg  = tid & 3;           // 16B segment within a 64B K-row
  int bcopycol = n0 + crow0;
  if (bcopycol >= N) bcopycol = N - 1; // clamp; junk cols masked at store
  const _Float16* gA0 = A + (size_t)(m0 + crow0)      * K + cseg * 8;
  const _Float16* gA1 = A + (size_t)(m0 + crow0 + 64) * K + cseg * 8;
  const _Float16* gB  = W + (size_t)bcopycol          * K + cseg * 8;
  const unsigned lA0 = (unsigned)(crow0        * LDS_ROW + cseg * 16);
  const unsigned lA1 = (unsigned)((crow0 + 64) * LDS_ROW + cseg * 16);
  const unsigned lB  = (unsigned)(ABUF_BYTES + crow0 * LDS_ROW + cseg * 16);

  auto issue_tile = [&](int k0, int buf) {
    char* base = smem + (unsigned)buf * BUF_BYTES;
    async_g2l_b128(base + lA0, gA0 + k0);
    async_g2l_b128(base + lA1, gA1 + k0);
    async_g2l_b128(base + lB,  gB  + k0);
  };

  // ---- per-wave LDS fragment offsets
  const unsigned aoff = (unsigned)((wave * 16 + l15) * LDS_ROW + ((lane < 16) ? 0 : 32));
  unsigned boff[4];
#pragma unroll
  for (int j = 0; j < 4; ++j)
    boff[j] = (unsigned)(ABUF_BYTES + (j * 16 + l15) * LDS_ROW + ((lane < 16) ? 0 : 32));

  v8f acc[4];
#pragma unroll
  for (int j = 0; j < 4; ++j)
#pragma unroll
    for (int r = 0; r < 8; ++r) acc[j][r] = 0.f;

  issue_tile(0, 0);
  wait_async0();
  __syncthreads();

  for (int k0 = 0; k0 < K; k0 += KSTEP) {
    const int cur = (k0 / KSTEP) & 1;
    if (k0 + KSTEP < K) issue_tile(k0 + KSTEP, cur ^ 1);  // overlap next-tile DMA

    const char* sb = smem + (unsigned)cur * BUF_BYTES;
    union { v16h v; u32x4 q[2]; } af;
    af.q[0] = *(const u32x4*)(sb + aoff);
    af.q[1] = *(const u32x4*)(sb + aoff + 16);
#pragma unroll
    for (int j = 0; j < 4; ++j) {
      union { v16h v; u32x4 q[2]; } bf;
      bf.q[0] = *(const u32x4*)(sb + boff[j]);
      bf.q[1] = *(const u32x4*)(sb + boff[j] + 16);
      // (neg_a, A, neg_b, B, c_mod, C, reuse_a, reuse_b)
      acc[j] = __builtin_amdgcn_wmma_f32_16x16x32_f16(
          false, af.v, false, bf.v, (short)0, acc[j], false, false);
    }
    wait_async0();
    __syncthreads();
  }

  const int crow = m0 + wave * 16 + ((lane < 16) ? 0 : 8);
#pragma unroll
  for (int j = 0; j < 4; ++j) {
    const int ccol = n0 + j * 16 + l15;
    if (ccol < N) {
      float* Cp = C + (size_t)crow * N + ccol;
#pragma unroll
      for (int r = 0; r < 8; ++r) Cp[(size_t)r * N] = acc[j][r];
    }
  }
}

// ---------------------------------------------------------------------------
// Depthwise causal conv (K=4) + bias + SiLU over the xBC slice, plus
// dt = softplus(zx[...,2176+h] + dt_bias). Outputs stored in *scan time*.
// ---------------------------------------------------------------------------
__global__ void conv_dt_kernel(const float* __restrict__ zx,
                               const float* __restrict__ Wc,
                               const float* __restrict__ bc,
                               const float* __restrict__ dtb,
                               float* __restrict__ xbc,
                               float* __restrict__ dtbuf, int dir) {
  const int CPT = CONV_DIM + NHEADS;
  int idx = blockIdx.x * 256 + threadIdx.x;
  if (idx >= NROWS * CPT) return;
  int c = idx % CPT;
  int row = idx / CPT;
  int b = row >> 10;
  int t = row & 1023;
  if (c < CONV_DIM) {
    float acc = 0.f;
#pragma unroll
    for (int k = 0; k < K_CONV; ++k) {
      int u = t - (K_CONV - 1) + k;
      if (u >= 0) {
        int s = dir ? (SEQLEN - 1 - u) : u;
        acc += Wc[c * K_CONV + k] *
               zx[((size_t)(b * SEQLEN + s)) * D_IN_PROJ + D_INNER + c];
      }
    }
    acc += bc[c];
    xbc[(size_t)row * CONV_DIM + c] = acc / (1.f + expf(-acc));  // SiLU
  } else {
    int hd = c - CONV_DIM;
    int s = dir ? (SEQLEN - 1 - t) : t;
    float v = zx[((size_t)(b * SEQLEN + s)) * D_IN_PROJ + (D_INNER + CONV_DIM) + hd] + dtb[hd];
    dtbuf[row * NHEADS + hd] = (v > 20.f) ? v : log1pf(expf(v));  // softplus
  }
}

// ---------------------------------------------------------------------------
// Sequential SSM scan. One workgroup per (batch, head); 64x64 fp32 state in
// registers (16 per thread). Thread (p = tid>>2, q = tid&3): row p, cols
// q*16..+15. B/C/x via LDS; y reduced across 4 lanes with wave32 shuffles.
// ---------------------------------------------------------------------------
__global__ void ssm_scan_kernel(const float* __restrict__ xbc,
                                const float* __restrict__ dtbuf,
                                const float* __restrict__ Alog,
                                const float* __restrict__ Dp,
                                float* __restrict__ y) {
  const int b = blockIdx.x >> 4;
  const int hd = blockIdx.x & 15;
  const int tid = threadIdx.x;
  const int p = tid >> 2;
  const int q = tid & 3;
  __shared__ float sX[HEADDIM], sB[D_STATE], sC[D_STATE];
  float hstate[16];
#pragma unroll
  for (int i = 0; i < 16; ++i) hstate[i] = 0.f;
  const float Av = -expf(Alog[hd]);
  const float Dv = Dp[hd];
  for (int t = 0; t < SEQLEN; ++t) {
    const size_t roff = ((size_t)(b * SEQLEN + t)) * CONV_DIM;
    if (tid < 64)        sX[tid]       = xbc[roff + hd * HEADDIM + tid];
    else if (tid < 128)  sB[tid - 64]  = xbc[roff + D_INNER + (tid - 64)];
    else if (tid < 192)  sC[tid - 128] = xbc[roff + D_INNER + D_STATE + (tid - 128)];
    __syncthreads();
    const float dtv = dtbuf[(b * SEQLEN + t) * NHEADS + hd];
    const float dAv = expf(dtv * Av);
    const float xp = sX[p];
    float yp = 0.f;
#pragma unroll
    for (int i = 0; i < 16; ++i) {
      const int n = (q << 4) + i;
      hstate[i] = hstate[i] * dAv + dtv * sB[n] * xp;
      yp += hstate[i] * sC[n];
    }
    yp += __shfl_xor(yp, 1, 32);
    yp += __shfl_xor(yp, 2, 32);
    if (q == 0)
      y[((size_t)(b * SEQLEN + t)) * D_INNER + hd * HEADDIM + p] = yp + Dv * xp;
    __syncthreads();
  }
}

// ---------------------------------------------------------------------------
// y = rmsnorm(y * silu(z), gw, 1e-5) -> f16 for out-proj GEMM.
// ---------------------------------------------------------------------------
__global__ void gate_norm_kernel(const float* __restrict__ y_ssm,
                                 const float* __restrict__ zx,
                                 const float* __restrict__ gw,
                                 _Float16* __restrict__ yn, int dir) {
  const int row = blockIdx.x;           // b*SEQLEN + t_scan
  const int b = row >> 10;
  const int t = row & 1023;
  const int s = dir ? (SEQLEN - 1 - t) : t;
  const float* zrow = zx + ((size_t)(b * SEQLEN + s)) * D_IN_PROJ;
  const float* yrow = y_ssm + (size_t)row * D_INNER;
  float g[4];
  float ss = 0.f;
#pragma unroll
  for (int i = 0; i < 4; ++i) {
    const int c = threadIdx.x + i * 256;
    const float z = zrow[c];
    const float v = yrow[c] * (z / (1.f + expf(-z)));
    g[i] = v;
    ss += v * v;
  }
  __shared__ float red[256];
  red[threadIdx.x] = ss;
  __syncthreads();
  for (int ofs = 128; ofs > 0; ofs >>= 1) {
    if (threadIdx.x < ofs) red[threadIdx.x] += red[threadIdx.x + ofs];
    __syncthreads();
  }
  const float rms = rsqrtf(red[0] / (float)D_INNER + 1e-5f);
#pragma unroll
  for (int i = 0; i < 4; ++i) {
    const int c = threadIdx.x + i * 256;
    yn[(size_t)row * D_INNER + c] = (_Float16)(g[i] * rms * gw[c]);
  }
}

// ---------------------------------------------------------------------------
// Concatenate fwd and (un-flipped) bwd outputs to f16 for the combine GEMM.
// ---------------------------------------------------------------------------
__global__ void cat_kernel(const float* __restrict__ df,
                           const float* __restrict__ db,
                           _Float16* __restrict__ cat) {
  int idx = blockIdx.x * 256 + threadIdx.x;
  if (idx >= NROWS * 2 * D_MODEL) return;
  int row = idx >> 10;
  int c = idx & 1023;
  int b = row >> 10;
  int t = row & 1023;
  float v;
  if (c < D_MODEL)
    v = df[(size_t)row * D_MODEL + c];
  else
    v = db[((size_t)(b * SEQLEN + (SEQLEN - 1 - t))) * D_MODEL + (c - D_MODEL)];
  cat[idx] = (_Float16)v;
}

// ---------------------------------------------------------------------------
// h = rmsnorm(h + combined, norm_w, finfo(f32).eps)
// ---------------------------------------------------------------------------
__global__ void resid_norm_kernel(const float* __restrict__ hin,
                                  const float* __restrict__ comb,
                                  const float* __restrict__ nw,
                                  float* __restrict__ hout,
                                  float* __restrict__ finalout) {
  const int row = blockIdx.x;
  float v[2];
  float ss = 0.f;
#pragma unroll
  for (int i = 0; i < 2; ++i) {
    const int c = threadIdx.x + i * 256;
    v[i] = hin[(size_t)row * D_MODEL + c] + comb[(size_t)row * D_MODEL + c];
    ss += v[i] * v[i];
  }
  __shared__ float red[256];
  red[threadIdx.x] = ss;
  __syncthreads();
  for (int ofs = 128; ofs > 0; ofs >>= 1) {
    if (threadIdx.x < ofs) red[threadIdx.x] += red[threadIdx.x + ofs];
    __syncthreads();
  }
  const float rms = rsqrtf(red[0] / (float)D_MODEL + 1.1920929e-07f);
#pragma unroll
  for (int i = 0; i < 2; ++i) {
    const int c = threadIdx.x + i * 256;
    const float o = v[i] * rms * nw[c];
    hout[(size_t)row * D_MODEL + c] = o;
    if (finalout) finalout[(size_t)row * D_MODEL + c] = o;
  }
}

// ---------------------------------------------------------------------------
// Host orchestration
// ---------------------------------------------------------------------------
static inline void launch_cvt(const float* s, _Float16* d, int n, hipStream_t st) {
  cvt_f32_f16_kernel<<<(n + 255) / 256, 256, 0, st>>>(s, d, n);
}
static inline void launch_gemm(const _Float16* A, const _Float16* W, float* C,
                               int M, int N, int K, hipStream_t st) {
  int tilesN = (N + NBLK - 1) / NBLK;
  int blocks = (M / MBLK) * tilesN;
  wmma_gemm_f16<<<blocks, 256, 0, st>>>(A, W, C, M, N, K);
}

extern "C" void kernel_launch(void* const* d_in, const int* in_sizes, int n_in,
                              void* d_out, int out_size, void* d_ws, size_t ws_size,
                              hipStream_t stream) {
  (void)in_sizes; (void)n_in; (void)out_size; (void)ws_size;
  const float* x         = (const float*)d_in[0];
  const float* in_proj_w = (const float*)d_in[1];   // [4, 2192, 512]
  const float* conv_w    = (const float*)d_in[2];   // [4, 1152, 4]
  const float* conv_b    = (const float*)d_in[3];   // [4, 1152]
  const float* dt_bias   = (const float*)d_in[4];   // [4, 16]
  const float* A_log     = (const float*)d_in[5];   // [4, 16]
  const float* D_skip    = (const float*)d_in[6];   // [4, 16]
  const float* gnorm_w   = (const float*)d_in[7];   // [4, 1024]
  const float* mout_w    = (const float*)d_in[8];   // [4, 512, 1024]
  const float* norm_w    = (const float*)d_in[9];   // [2, 512]
  const float* outproj_w = (const float*)d_in[10];  // [512, 1024]

  char* ws = (char*)d_ws;
  size_t o = 0;
  auto take = [&](size_t bytes) { char* p = ws + o; o += (bytes + 255) & ~(size_t)255; return p; };
  float*    h_f32   = (float*)   take((size_t)NROWS * D_MODEL * 4);
  _Float16* h_f16   = (_Float16*)take((size_t)NROWS * D_MODEL * 2);
  float*    zx      = (float*)   take((size_t)NROWS * D_IN_PROJ * 4);
  float*    xbc     = (float*)   take((size_t)NROWS * CONV_DIM * 4);
  float*    dtbuf   = (float*)   take((size_t)NROWS * NHEADS * 4);
  float*    y_ssm   = (float*)   take((size_t)NROWS * D_INNER * 4);
  _Float16* yn_f16  = (_Float16*)take((size_t)NROWS * D_INNER * 2);
  float*    dir_f   = (float*)   take((size_t)NROWS * D_MODEL * 4);
  float*    dir_b   = (float*)   take((size_t)NROWS * D_MODEL * 4);
  _Float16* cat_f16 = (_Float16*)take((size_t)NROWS * 2 * D_MODEL * 2);
  float*    comb    = (float*)   take((size_t)NROWS * D_MODEL * 4);
  _Float16* wA_f16  = (_Float16*)take((size_t)D_IN_PROJ * D_MODEL * 2);
  _Float16* wB_f16  = (_Float16*)take((size_t)D_MODEL * D_INNER * 2);

  hipMemcpyAsync(h_f32, x, (size_t)NROWS * D_MODEL * sizeof(float),
                 hipMemcpyDeviceToDevice, stream);

  for (int l = 0; l < 2; ++l) {
    launch_cvt(h_f32, h_f16, NROWS * D_MODEL, stream);
    for (int d2 = 0; d2 < 2; ++d2) {                  // 0 = fwd, 1 = bwd
      const int d = 2 * l + d2;
      launch_cvt(in_proj_w + (size_t)d * D_IN_PROJ * D_MODEL, wA_f16,
                 D_IN_PROJ * D_MODEL, stream);
      launch_gemm(h_f16, wA_f16, zx, NROWS, D_IN_PROJ, D_MODEL, stream);
      {
        int total = NROWS * (CONV_DIM + NHEADS);
        conv_dt_kernel<<<(total + 255) / 256, 256, 0, stream>>>(
            zx, conv_w + (size_t)d * CONV_DIM * K_CONV, conv_b + (size_t)d * CONV_DIM,
            dt_bias + d * NHEADS, xbc, dtbuf, d2);
      }
      ssm_scan_kernel<<<BATCH * NHEADS, 256, 0, stream>>>(
          xbc, dtbuf, A_log + d * NHEADS, D_skip + d * NHEADS, y_ssm);
      gate_norm_kernel<<<NROWS, 256, 0, stream>>>(
          y_ssm, zx, gnorm_w + (size_t)d * D_INNER, yn_f16, d2);
      launch_cvt(mout_w + (size_t)d * D_MODEL * D_INNER, wB_f16,
                 D_MODEL * D_INNER, stream);
      launch_gemm(yn_f16, wB_f16, d2 ? dir_b : dir_f, NROWS, D_MODEL, D_INNER, stream);
    }
    {
      int total = NROWS * 2 * D_MODEL;
      cat_kernel<<<(total + 255) / 256, 256, 0, stream>>>(dir_f, dir_b, cat_f16);
    }
    launch_cvt(outproj_w, wB_f16, D_MODEL * 2 * D_MODEL, stream);
    launch_gemm(cat_f16, wB_f16, comb, NROWS, D_MODEL, 2 * D_MODEL, stream);
    resid_norm_kernel<<<NROWS, 256, 0, stream>>>(
        h_f32, comb, norm_w + (size_t)l * D_MODEL, h_f32,
        (l == 1) ? (float*)d_out : (float*)nullptr);
  }
}
